// TransformerTranslator_17325898072659
// MI455X (gfx1250) — matrix operand order repaired and checked by reading, not compiled, not run on hardware
//
#include <hip/hip_runtime.h>
#include <hip/hip_bf16.h>
#include <math.h>

// Problem constants (match reference)
#define NB 16
#define TT 2048
#define HH 128
#define DKK 96
#define QSCALE 0.1020620726159658f   // 1/sqrt(96)

typedef __attribute__((ext_vector_type(16))) __bf16 v16bf;
typedef __attribute__((ext_vector_type(8)))  float  v8f;

union BF16x16 { unsigned int u[8]; v16bf v; };

#define WMMA_BF16(A, B, C) \
  __builtin_amdgcn_wmma_f32_16x16x32_bf16(false, (A), false, (B), (short)0, (C), false, false)

// ---- A fragment: 16x32 (MxK) tile, row-major, leading dim ld (elements) ----
// ISA 7.12.2: lane l holds row M=l&15; VGPR i(0..3): K = kb+2i,+1 (kb = l>=16 ? 8:0);
// VGPR i(4..7): K = 16+kb+2(i-4),+1.  Adjacent K pairs -> aligned dword loads.
__device__ __forceinline__ v16bf load_a_frag(const __bf16* base, int ld) {
  int lane = threadIdx.x & 31;
  int row  = lane & 15;
  int kb   = (lane & 16) ? 8 : 0;
  const __bf16* r0 = base + (size_t)row * ld;
  BF16x16 a;
#pragma unroll
  for (int i = 0; i < 4; ++i)
    a.u[i]     = *(const unsigned int*)(r0 + kb + 2 * i);
#pragma unroll
  for (int i = 0; i < 4; ++i)
    a.u[4 + i] = *(const unsigned int*)(r0 + 16 + kb + 2 * i);
  return a.v;
}

// ---- B fragment: 32x16 (KxN) tile where element (k,n) lives at base[n*ld + k] ----
// ISA layout: lane l holds column N=l&15; koff = l>=16 ? 8:0; VGPR j(0..3): K = koff+2j,+1;
// VGPR j(4..7): K = 16+koff+2(j-4),+1.  Adjacent K pairs -> aligned dword loads.
__device__ __forceinline__ v16bf load_b_frag(const __bf16* base, int ld) {
  int lane = threadIdx.x & 31;
  int n    = lane & 15;
  int koff = (lane & 16) ? 8 : 0;
  const __bf16* c = base + (size_t)n * ld + koff;
  BF16x16 b;
#pragma unroll
  for (int j = 0; j < 8; ++j)
    b.u[j] = *(const unsigned int*)(c + (j < 4 ? 2 * j : 16 + 2 * (j - 4)));
  return b.v;
}

// Row reductions within 16-lane halves (C/D layout: VGPR r = row r (lanes 0-15) or r+8 (16-31))
__device__ __forceinline__ float rmax16(float v) {
#pragma unroll
  for (int m = 1; m <= 8; m <<= 1) v = fmaxf(v, __shfl_xor(v, m, 32));
  return v;
}
__device__ __forceinline__ float rsum16(float v) {
#pragma unroll
  for (int m = 1; m <= 8; m <<= 1) v += __shfl_xor(v, m, 32);
  return v;
}

// CDNA5 async global->LDS copy (GV mode), tracked by ASYNCcnt.
// lds_off: LDS byte address (low 32 bits of the generic/flat shared-aperture address).
__device__ __forceinline__ void async_copy_b128(unsigned lds_off, const void* gaddr) {
  asm volatile("global_load_async_to_lds_b128 %0, %1, off"
               :: "v"(lds_off), "v"(gaddr) : "memory");
}
__device__ __forceinline__ void wait_async0() {
  asm volatile("s_wait_asynccnt 0x0" ::: "memory");
}
__device__ __forceinline__ unsigned lds_addr32(const void* p) {
  return (unsigned)(size_t)p;   // LDS flat aperture keeps the offset in addr[31:0]
}

// ---------------- K0: f32 -> bf16 weight conversion ----------------
__global__ void k_f32_to_bf16(const float* __restrict__ s, __bf16* __restrict__ d, int n) {
  int i = blockIdx.x * blockDim.x + threadIdx.x;
  if (i < n) d[i] = (__bf16)s[i];
}

// ---------------- K1: embedding + positional encoding ----------------
__global__ void k_embed(const int* __restrict__ inp, const float* __restrict__ Wemb,
                        float* __restrict__ embf, __bf16* __restrict__ embb) {
  int i  = blockIdx.x * blockDim.x + threadIdx.x;   // over NB*TT*HH
  int h  = i & (HH - 1);
  int nt = i >> 7;
  int t  = nt & (TT - 1);
  int tok = inp[nt];
  float v = Wemb[(size_t)tok * HH + h] * 0.25f;     // / sqrt(N=16)
  float e = (float)(h & ~1) * (1.0f / (float)HH);
  float ang = (float)t * __powf(10000.0f, -e);
  v += (h & 1) ? __cosf(ang) : __sinf(ang);
  embf[i] = v;
  embb[i] = (__bf16)v;
}

// ---------------- K2: all six QKV projections via WMMA ----------------
struct QkvArgs {
  const __bf16* emb;
  const __bf16* W[6];   // bf16 weights [DK,H] row-major
  const float*  b[6];
  __bf16*       out[6]; // q,k row-major [N,T,DK]; v stored transposed [N,DK,T]
};

__global__ void k_qkv(QkvArgs a) {
  int wid = blockIdx.x * (blockDim.x >> 5) + (threadIdx.x >> 5);
  int p    = wid / 12288;           // 6 projections
  int rem  = wid % 12288;
  int nb   = rem / 768;
  int rem2 = rem % 768;
  int mt   = rem2 / 6, ct = rem2 % 6;
  int m0 = mt * 16, n0 = ct * 16;

  const __bf16* A = a.emb + (size_t)(nb * TT + m0) * HH;
  const __bf16* W = a.W[p] + (size_t)n0 * HH;
  v8f acc = {};
#pragma unroll
  for (int ks = 0; ks < 4; ++ks) {                  // K = 128 = 4 x 32
    v16bf af = load_a_frag(A + ks * 32, HH);
    v16bf bf = load_b_frag(W + ks * 32, HH);
    acc = WMMA_BF16(af, bf, acc);
  }
  int lane = threadIdx.x & 31;
  int col  = lane & 15;
  int rb   = (lane & 16) ? 8 : 0;
  int d    = n0 + col;
  float bias  = a.b[p][d];
  float scale = (p == 0 || p == 3) ? QSCALE : 1.0f; // fold 1/sqrt(DK) into q
  bool  isV   = (p == 2 || p == 5);
#pragma unroll
  for (int r = 0; r < 8; ++r) {
    int row = m0 + r + rb;
    float val = (acc[r] + bias) * scale;
    if (isV) a.out[p][(size_t)(nb * DKK + d) * TT + row] = (__bf16)val;   // V^T
    else     a.out[p][(size_t)(nb * TT + row) * DKK + d] = (__bf16)val;
  }
}

// ---------------- K3: flash attention ----------------
// One 128-thread block = 128 query rows (4 waves x 2 row-tiles of 16).
// K/V tiles (32 keys) staged in LDS via async copies, double-buffered, shared by
// all 4 waves; every K/V B-fragment feeds two WMMAs (two row-tiles per wave).
struct AttnArgs {
  const __bf16* q;    // [N,T,DK] (pre-scaled)
  const __bf16* k;    // [N,T,DK]
  const __bf16* vT;   // [N,DK,T]
  __bf16*       cat;  // [N,T,2*DK]
  int           dOff; // head2 -> 0, head1 -> DK (concat order [h2,h1])
};

__global__ void k_attn(AttnArgs a) {
  __shared__ __bf16 kbuf[2][32 * DKK];      // [key][d]   6 KB per buffer
  __shared__ __bf16 vbuf[2][DKK * 32];      // [d][key]   6 KB per buffer
  __shared__ __bf16 plds[4][2][16 * 32];    // per-wave, per-row-tile P staging

  int tid  = threadIdx.x;
  int wv   = tid >> 5;
  int lane = tid & 31;
  int col  = lane & 15;
  int rb   = (lane & 16) ? 8 : 0;

  int nb   = blockIdx.x >> 4;               // T/128 = 16 blocks per batch
  int m0   = (blockIdx.x & 15) * 128 + wv * 32;

  const __bf16* Kb = a.k  + (size_t)nb * TT * DKK;          // rows contiguous (192B)
  const __bf16* Vb = a.vT + (size_t)nb * DKK * TT;          // row stride 4KB, 64B used

  // ---- load Q fragments for both row-tiles ----
  v16bf qf[2][3];
#pragma unroll
  for (int rt = 0; rt < 2; ++rt)
#pragma unroll
    for (int ks = 0; ks < 3; ++ks)
      qf[rt][ks] = load_a_frag(a.q + (size_t)(nb * TT + m0 + rt * 16) * DKK + ks * 32, DKK);

  float mrow[2][8], lsum[2][8];
  v8f O[2][6];
#pragma unroll
  for (int rt = 0; rt < 2; ++rt) {
#pragma unroll
    for (int r = 0; r < 8; ++r) { mrow[rt][r] = -1e30f; lsum[rt][r] = 0.0f; }
#pragma unroll
    for (int ct = 0; ct < 6; ++ct) O[rt][ct] = (v8f){};
  }

  // ---- cooperative async stage of one 32-key K/V tile into LDS buffer `buf` ----
  // K tile: 32 rows x 192B, globally contiguous -> 384 b128 chunks.
  // V tile: 96 rows x 64B (row stride 4KB)      -> 384 b128 chunks.
  auto stage = [&](int buf, int s0) {
#pragma unroll
    for (int j = 0; j < 3; ++j) {
      int c = tid + j * 128;                // 0..383
      async_copy_b128(lds_addr32((char*)&kbuf[buf][0] + c * 16),
                      (const char*)(Kb + (size_t)s0 * DKK) + c * 16);
      int vr = c >> 2, vs = c & 3;
      async_copy_b128(lds_addr32((char*)&vbuf[buf][0] + c * 16),
                      (const char*)(Vb + (size_t)vr * TT + s0) + vs * 16);
    }
  };

  stage(0, 0);
  wait_async0();
  __syncthreads();

  for (int it = 0; it < TT / 32; ++it) {
    int s0  = it * 32;
    int cur = it & 1;
    if (it + 1 < TT / 32) stage(cur ^ 1, s0 + 32);   // overlap next tile with compute

    // ---- S = Q K^T : two key-tiles x three K-steps, B-frags shared across row-tiles ----
    v8f sacc[2][2] = {{{}, {}}, {{}, {}}};
#pragma unroll
    for (int kt = 0; kt < 2; ++kt)
#pragma unroll
      for (int ks = 0; ks < 3; ++ks) {
        v16bf b = load_b_frag(&kbuf[cur][0] + (size_t)(kt * 16) * DKK + ks * 32, DKK);
#pragma unroll
        for (int rt = 0; rt < 2; ++rt)
          sacc[rt][kt] = WMMA_BF16(qf[rt][ks], b, sacc[rt][kt]);
      }

    // ---- online softmax per row-tile; P -> LDS (bf16) ----
    float alpha[2][8];
#pragma unroll
    for (int rt = 0; rt < 2; ++rt)
#pragma unroll
      for (int r = 0; r < 8; ++r) {
        float mn = fmaxf(mrow[rt][r], rmax16(fmaxf(sacc[rt][0][r], sacc[rt][1][r])));
        float p0 = __expf(sacc[rt][0][r] - mn);
        float p1 = __expf(sacc[rt][1][r] - mn);
        alpha[rt][r] = __expf(mrow[rt][r] - mn);
        mrow[rt][r]  = mn;
        lsum[rt][r]  = lsum[rt][r] * alpha[rt][r] + rsum16(p0 + p1);
        plds[wv][rt][(r + rb) * 32 + col]      = (__bf16)p0;
        plds[wv][rt][(r + rb) * 32 + col + 16] = (__bf16)p1;
      }
    asm volatile("s_wait_dscnt 0x0" ::: "memory");   // per-wave LDS RAW on P
    v16bf pf[2];
    pf[0] = load_a_frag(&plds[wv][0][0], 32);
    pf[1] = load_a_frag(&plds[wv][1][0], 32);

    // ---- O = O*alpha + P V : V B-frags shared across row-tiles ----
#pragma unroll
    for (int ct = 0; ct < 6; ++ct) {
      v16bf vb = load_b_frag(&vbuf[cur][0] + (size_t)(ct * 16) * 32, 32);
#pragma unroll
      for (int rt = 0; rt < 2; ++rt) {
#pragma unroll
        for (int r = 0; r < 8; ++r) O[rt][ct][r] *= alpha[rt][r];
        O[rt][ct] = WMMA_BF16(pf[rt], vb, O[rt][ct]);
      }
    }

    wait_async0();        // next tile landed in LDS
    __syncthreads();      // all waves done reading current tile -> safe to overwrite
  }

  // ---- normalize and write into concat buffer ----
#pragma unroll
  for (int rt = 0; rt < 2; ++rt)
#pragma unroll
    for (int r = 0; r < 8; ++r) {
      float inv = 1.0f / lsum[rt][r];
      int row = m0 + rt * 16 + r + rb;
#pragma unroll
      for (int ct = 0; ct < 6; ++ct) {
        int d = ct * 16 + col;
        a.cat[(size_t)(nb * TT + row) * (2 * DKK) + a.dOff + d] = (__bf16)(O[rt][ct][r] * inv);
      }
    }
}

// ---------------- K4: output projection + residual + LayerNorm ----------------
__global__ void k_proj_ln(const __bf16* __restrict__ cat, const __bf16* __restrict__ wp,
                          const float* __restrict__ bp, const float* __restrict__ embf,
                          const float* __restrict__ gamma, const float* __restrict__ beta,
                          float* __restrict__ out) {
  int wid  = blockIdx.x * 4 + (threadIdx.x >> 5);   // NB*TT/16 waves
  int nb   = wid >> 7;
  int m0   = (wid & 127) * 16;
  int lane = threadIdx.x & 31;
  int col  = lane & 15;
  int rb   = (lane & 16) ? 8 : 0;

  v16bf af[6];
#pragma unroll
  for (int ks = 0; ks < 6; ++ks)                    // K = 192 = 6 x 32
    af[ks] = load_a_frag(cat + (size_t)(nb * TT + m0) * (2 * DKK) + ks * 32, 2 * DKK);

  float x[8][8];
#pragma unroll
  for (int ctile = 0; ctile < 8; ++ctile) {         // H = 128 = 8 column tiles
    v8f acc = {};
#pragma unroll
    for (int ks = 0; ks < 6; ++ks) {
      v16bf b = load_b_frag(wp + (size_t)(ctile * 16) * (2 * DKK) + ks * 32, 2 * DKK);
      acc = WMMA_BF16(af[ks], b, acc);
    }
    int h = ctile * 16 + col;
#pragma unroll
    for (int r = 0; r < 8; ++r) {
      int row = m0 + r + rb;
      x[ctile][r] = acc[r] + bp[h] + embf[(size_t)(nb * TT + row) * HH + h];
    }
  }
#pragma unroll
  for (int r = 0; r < 8; ++r) {
    float s = 0.0f;
#pragma unroll
    for (int ct = 0; ct < 8; ++ct) s += x[ct][r];
    float mu = rsum16(s) * (1.0f / (float)HH);
    float s2 = 0.0f;
#pragma unroll
    for (int ct = 0; ct < 8; ++ct) { float d = x[ct][r] - mu; s2 += d * d; }
    float rstd = rsqrtf(rsum16(s2) * (1.0f / (float)HH) + 1e-5f);
    int row = m0 + r + rb;
#pragma unroll
    for (int ct = 0; ct < 8; ++ct) {
      int h = ct * 16 + col;
      out[(size_t)(nb * TT + row) * HH + h] = gamma[h] * (x[ct][r] - mu) * rstd + beta[h];
    }
  }
}

// ---------------- host-side orchestration ----------------
extern "C" void kernel_launch(void* const* d_in, const int* in_sizes, int n_in,
                              void* d_out, int out_size, void* d_ws, size_t ws_size,
                              hipStream_t stream) {
  (void)in_sizes; (void)n_in; (void)out_size; (void)ws_size;
  const int*   inputs = (const int*)d_in[0];
  const float* Wemb   = (const float*)d_in[1];
  const float* Wsrc[7] = { (const float*)d_in[2], (const float*)d_in[4], (const float*)d_in[6],
                           (const float*)d_in[8], (const float*)d_in[10], (const float*)d_in[12],
                           (const float*)d_in[14] };                     // Wq1,Wk1,Wv1,Wq2,Wk2,Wv2,Wp
  const float* bq1 = (const float*)d_in[3],  *bk1 = (const float*)d_in[5],
             * bv1 = (const float*)d_in[7],  *bq2 = (const float*)d_in[9],
             * bk2 = (const float*)d_in[11], *bv2 = (const float*)d_in[13],
             * bp  = (const float*)d_in[15], *gamma = (const float*)d_in[16],
             * beta = (const float*)d_in[17];

  char* ws = (char*)d_ws;
  size_t off = 0;
  auto alloc = [&](size_t bytes) { size_t o = off; off = (off + bytes + 255) & ~(size_t)255; return o; };

  float*  embf = (float*)(ws + alloc((size_t)NB * TT * HH * 4));
  __bf16* embb = (__bf16*)(ws + alloc((size_t)NB * TT * HH * 2));
  __bf16* Wbf[7];
  for (int i = 0; i < 6; ++i) Wbf[i] = (__bf16*)(ws + alloc((size_t)DKK * HH * 2));
  Wbf[6] = (__bf16*)(ws + alloc((size_t)HH * 2 * DKK * 2));
  __bf16* q1  = (__bf16*)(ws + alloc((size_t)NB * TT * DKK * 2));
  __bf16* k1  = (__bf16*)(ws + alloc((size_t)NB * TT * DKK * 2));
  __bf16* vT1 = (__bf16*)(ws + alloc((size_t)NB * DKK * TT * 2));
  __bf16* q2  = (__bf16*)(ws + alloc((size_t)NB * TT * DKK * 2));
  __bf16* k2  = (__bf16*)(ws + alloc((size_t)NB * TT * DKK * 2));
  __bf16* vT2 = (__bf16*)(ws + alloc((size_t)NB * DKK * TT * 2));
  __bf16* cat = (__bf16*)(ws + alloc((size_t)NB * TT * 2 * DKK * 2));

  // weight conversion
  for (int i = 0; i < 6; ++i)
    k_f32_to_bf16<<<(DKK * HH + 255) / 256, 256, 0, stream>>>(Wsrc[i], Wbf[i], DKK * HH);
  k_f32_to_bf16<<<(HH * 2 * DKK + 255) / 256, 256, 0, stream>>>(Wsrc[6], Wbf[6], HH * 2 * DKK);

  // embedding + positional encoding
  k_embed<<<(NB * TT * HH) / 256, 256, 0, stream>>>(inputs, Wemb, embf, embb);

  // QKV projections (6 proj x 16 batch x 128 row-tiles x 6 col-tiles = 73728 waves)
  QkvArgs qa;
  qa.emb = embb;
  qa.W[0] = Wbf[0]; qa.W[1] = Wbf[1]; qa.W[2] = Wbf[2];
  qa.W[3] = Wbf[3]; qa.W[4] = Wbf[4]; qa.W[5] = Wbf[5];
  qa.b[0] = bq1; qa.b[1] = bk1; qa.b[2] = bv1;
  qa.b[3] = bq2; qa.b[4] = bk2; qa.b[5] = bv2;
  qa.out[0] = q1; qa.out[1] = k1; qa.out[2] = vT1;
  qa.out[3] = q2; qa.out[4] = k2; qa.out[5] = vT2;
  k_qkv<<<73728 / 4, 128, 0, stream>>>(qa);

  // attention heads (concat order: [h2, h1]); 16 batches x 16 row-blocks = 256 blocks/head
  AttnArgs a1{q1, k1, vT1, cat, DKK};
  AttnArgs a2{q2, k2, vT2, cat, 0};
  k_attn<<<NB * (TT / 128), 128, 0, stream>>>(a1);
  k_attn<<<NB * (TT / 128), 128, 0, stream>>>(a2);

  // projection + residual + layernorm
  k_proj_ln<<<(NB * TT / 16) / 4, 128, 0, stream>>>(cat, Wbf[6], bp, embf, gamma, beta, (float*)d_out);
}
